// mse_cos_contrastive_loss_39496519254725
// MI455X (gfx1250) — matrix operand order repaired and checked by reading, not compiled, not run on hardware
//
#include <hip/hip_runtime.h>
#include <hip/hip_bf16.h>

// ---------------------------------------------------------------------------
// MI455X (gfx1250) fused MSE + cosine-embedding + NT-Xent contrastive loss.
// Heavy op: 8192x8192x1024 cosine-sim GEMM -> v_wmma_f32_16x16x32_bf16,
// fed by double-buffered GLOBAL_LOAD_ASYNC_TO_LDS_B128 DMA (ASYNCcnt).
// logsumexp folded per tile (sim bounded in [-2,2]: no max subtraction),
// so the 256MB sim matrix is never materialized. All reductions are
// deterministic (exclusive-slot LDS tables + fixed-order folds, no atomics).
// ---------------------------------------------------------------------------

typedef __attribute__((ext_vector_type(16))) __bf16       v16bf;
typedef __attribute__((ext_vector_type(8)))  float        v8f;
typedef __attribute__((ext_vector_type(8)))  unsigned int v8u;

constexpr int NROWS = 8192;
constexpr int DDIM  = 1024;
constexpr int BM = 128, BN = 128, BK = 32;
constexpr int LSTR  = 20;          // LDS row stride in dwords (16 data + 4 pad)
constexpr int NBLK  = NROWS / BM;  // 64 tile-blocks per dimension
constexpr int NSTG  = DDIM / BK;   // 32 K-stages
constexpr float INV_T = 2.0f;      // 1 / TEMPERATURE

// Generic pointer -> 32-bit LDS byte offset (addrspacecast to LDS, truncate).
__device__ __forceinline__ unsigned lds_addr_of(const void* p) {
  return (unsigned)(unsigned long long)(const __attribute__((address_space(3))) char*)p;
}

// ---------------------------------------------------------------------------
// Stage 1: per-row norms, pair dot, squared-diff; emit normalized bf16 rows.
// One block (256 threads) per row; each thread owns one float4 (D/4 == 256).
// ---------------------------------------------------------------------------
__global__ __launch_bounds__(256) void stage1_rows(
    const float* __restrict__ yp, const float* __restrict__ yy,
    __bf16* __restrict__ abf, __bf16* __restrict__ bbf,
    float* __restrict__ cospair, float* __restrict__ msq) {
  const int row  = blockIdx.x;
  const int t    = threadIdx.x;
  const int lane = t & 31;
  const int wid  = t >> 5;

  const float4 p = ((const float4*)(yp + (size_t)row * DDIM))[t];
  const float4 q = ((const float4*)(yy + (size_t)row * DDIM))[t];

  float p2 = p.x*p.x + p.y*p.y + p.z*p.z + p.w*p.w;
  float y2 = q.x*q.x + q.y*q.y + q.z*q.z + q.w*q.w;
  float pd = p.x*q.x + p.y*q.y + p.z*q.z + p.w*q.w;
  float dx = p.x-q.x, dy = p.y-q.y, dz = p.z-q.z, dw = p.w-q.w;
  float sq = dx*dx + dy*dy + dz*dz + dw*dw;

  #pragma unroll
  for (int m = 16; m >= 1; m >>= 1) {
    p2 += __shfl_xor(p2, m, 32);
    y2 += __shfl_xor(y2, m, 32);
    pd += __shfl_xor(pd, m, 32);
    sq += __shfl_xor(sq, m, 32);
  }

  __shared__ float red[8][4];
  __shared__ float inv[2];
  if (lane == 0) { red[wid][0] = p2; red[wid][1] = y2; red[wid][2] = pd; red[wid][3] = sq; }
  __syncthreads();
  if (t == 0) {
    float P = 0.f, Y = 0.f, PD = 0.f, SQ = 0.f;
    #pragma unroll
    for (int w = 0; w < 8; ++w) { P += red[w][0]; Y += red[w][1]; PD += red[w][2]; SQ += red[w][3]; }
    float pn = fmaxf(sqrtf(P), 1e-8f);
    float yn = fmaxf(sqrtf(Y), 1e-8f);
    cospair[row] = PD / (pn * yn);
    msq[row]     = SQ;
    inv[0] = 1.0f / pn;
    inv[1] = 1.0f / yn;
  }
  __syncthreads();
  const float ip = inv[0], iy = inv[1];
  __bf16* ar = abf + (size_t)row * DDIM + t * 4;
  __bf16* br = bbf + (size_t)row * DDIM + t * 4;
  ar[0] = (__bf16)(p.x * ip); ar[1] = (__bf16)(p.y * ip);
  ar[2] = (__bf16)(p.z * ip); ar[3] = (__bf16)(p.w * ip);
  br[0] = (__bf16)(q.x * iy); br[1] = (__bf16)(q.y * iy);
  br[2] = (__bf16)(q.z * iy); br[3] = (__bf16)(q.w * iy);
}

// ---------------------------------------------------------------------------
// Stage 2: tiled bf16 WMMA GEMM sim = (A @ B^T) * (1/T) with on-the-fly
// exp-sum reduction. 256 threads = 8 waves (4m x 2n); each wave owns a
// 32x64 subtile (2x4 WMMA 16x16 accumulators). LDS tiles are double-
// buffered and filled by async DMA (global_load_async_to_lds_b128):
//   stage st: s_wait_asynccnt 0 ; barrier ; DMA stage st+1 -> buf[1-p] ;
//             compute from buf[p].
// Safe with a single barrier: every wave's fragment ds_loads of buf[1-p]
// (stage st-1) are s_wait_dscnt'ed before its WMMAs, hence complete before
// it reaches the barrier, so the new DMA cannot race them.
// ---------------------------------------------------------------------------
__global__ __launch_bounds__(256) void gemm_lse(
    const __bf16* __restrict__ A, const __bf16* __restrict__ B,
    float* __restrict__ rowpart, float* __restrict__ colpart) {
  __shared__ __align__(16) unsigned int AsB[2][BM * LSTR];  // 2 x 10 KB
  __shared__ __align__(16) unsigned int BsB[2][BN * LSTR];  // 2 x 10 KB
  __shared__ float rowsP[BM][32];   // [row][wn*16 + l16]   16 KB
  __shared__ float colsP[BN][8];    // [col][wm*2  + hi ]    4 KB

  const int bx = blockIdx.x;   // column tile of sim
  const int by = blockIdx.y;   // row tile of sim
  const int t    = threadIdx.x;
  const int lane = t & 31;
  const int wid  = t >> 5;
  const int wm   = wid >> 1;           // 0..3  (rows)
  const int wn   = wid & 1;            // 0..1  (cols)
  const bool hi  = lane >= 16;
  const int l16  = lane & 15;

  v8f acc[2][4];
  #pragma unroll
  for (int mi = 0; mi < 2; ++mi)
    #pragma unroll
    for (int ni = 0; ni < 4; ++ni)
      acc[mi][ni] = (v8f){0.f,0.f,0.f,0.f,0.f,0.f,0.f,0.f};

  // Async-DMA fill: thread t moves 32B of A-tile + 32B of B-tile per stage.
  // row = t>>1, 32-byte half selected by t&1. The immediate `offset:` is
  // added to BOTH the LDS and the global address (ISA 10.? async pseudocode),
  // so one LDS-addr VGPR + one 32-bit global-offset VGPR serve both b128s.
  const unsigned rowL  = (unsigned)(t >> 1);          // 0..127
  const unsigned halfB = (unsigned)(t & 1) * 32u;     // byte half of 64B row
  const unsigned gaBase = ((unsigned)(by * BM + (int)rowL) * (unsigned)DDIM) * 2u + halfB;
  const unsigned gbBase = ((unsigned)(bx * BN + (int)rowL) * (unsigned)DDIM) * 2u + halfB;
  const unsigned ldsRow = rowL * LSTR * 4u + halfB;   // byte offset within tile

  auto issue_fills = [&](int buf, int kb) {
    const unsigned la = lds_addr_of(&AsB[buf][0]) + ldsRow;
    const unsigned lb = lds_addr_of(&BsB[buf][0]) + ldsRow;
    const unsigned ga = gaBase + (unsigned)kb * 2u;
    const unsigned gb = gbBase + (unsigned)kb * 2u;
    asm volatile(
        "global_load_async_to_lds_b128 %0, %2, %4\n\t"
        "global_load_async_to_lds_b128 %0, %2, %4 offset:16\n\t"
        "global_load_async_to_lds_b128 %1, %3, %5\n\t"
        "global_load_async_to_lds_b128 %1, %3, %5 offset:16"
        :
        : "v"(la), "v"(lb), "v"(ga), "v"(gb), "s"(A), "s"(B)
        : "memory");
  };

  issue_fills(0, 0);   // prologue: stage 0 -> buffer 0

  for (int st = 0; st < NSTG; ++st) {
    const int p = st & 1;
    asm volatile("s_wait_asynccnt 0x0" ::: "memory");  // my buf[p] DMA done
    __syncthreads();                                   // everyone's done
    if (st + 1 < NSTG) issue_fills(1 - p, (st + 1) * BK);

    const unsigned int* __restrict__ As = AsB[p];
    const unsigned int* __restrict__ Bs = BsB[p];

    // B fragments: 32x16 bf16, V_v: lanes0-15 K=2v,2v+1 ; lanes16-31 K=16+2v
    v16bf bfr[4];
    #pragma unroll
    for (int ni = 0; ni < 4; ++ni) {
      const int n = wn * 64 + ni * 16 + l16;
      v8u u;
      #pragma unroll
      for (int v = 0; v < 8; ++v) u[v] = Bs[n * LSTR + v + (hi ? 8 : 0)];
      bfr[ni] = __builtin_bit_cast(v16bf, u);
    }
    // A fragments: 16x32 bf16, V_v: lanes0-15 K=2v,2v+1 ; lanes16-31 K=16+2v
    #pragma unroll
    for (int mi = 0; mi < 2; ++mi) {
      const int m = wm * 32 + mi * 16 + l16;
      v8u u;
      #pragma unroll
      for (int v = 0; v < 8; ++v) {
        const int pair = v + (hi ? 8 : 0);
        u[v] = As[m * LSTR + pair];
      }
      const v16bf af = __builtin_bit_cast(v16bf, u);
      #pragma unroll
      for (int ni = 0; ni < 4; ++ni)
        acc[mi][ni] = __builtin_amdgcn_wmma_f32_16x16x32_bf16(
            false, af, false, bfr[ni], (short)0, acc[mi][ni], false, false);
    }
  }

  // ---- epilogue: exp(sim) and shuffle-free deterministic reductions ----
  // C/D layout: lane l, VGPR v holds (m = v + 8*hi, n = l16) of each tile.
  // Row partial over this lane's 4 n-columns; col partial over its 16 rows.
  float rpart[2][8];
  float cpart[4] = {0.f, 0.f, 0.f, 0.f};
  #pragma unroll
  for (int mi = 0; mi < 2; ++mi)
    #pragma unroll
    for (int v = 0; v < 8; ++v) rpart[mi][v] = 0.f;

  #pragma unroll
  for (int mi = 0; mi < 2; ++mi)
    #pragma unroll
    for (int ni = 0; ni < 4; ++ni)
      #pragma unroll
      for (int v = 0; v < 8; ++v) {
        const float e = __expf(acc[mi][ni][v] * INV_T);
        rpart[mi][v] += e;
        cpart[ni]    += e;
      }

  // Exclusive-slot scatter: each LDS slot has exactly one writer.
  #pragma unroll
  for (int mi = 0; mi < 2; ++mi)
    #pragma unroll
    for (int v = 0; v < 8; ++v)
      rowsP[wm * 32 + mi * 16 + v + (hi ? 8 : 0)][wn * 16 + l16] = rpart[mi][v];
  #pragma unroll
  for (int ni = 0; ni < 4; ++ni)
    colsP[wn * 64 + ni * 16 + l16][wm * 2 + (hi ? 1 : 0)] = cpart[ni];
  __syncthreads();

  if (t < 128) {
    float r = 0.f;
    #pragma unroll
    for (int j = 0; j < 32; ++j) r += rowsP[t][j];
    rowpart[(size_t)bx * NROWS + by * BM + t] = r;
    float c = 0.f;
    #pragma unroll
    for (int j = 0; j < 8; ++j) c += colsP[t][j];
    colpart[(size_t)by * NROWS + bx * BN + t] = c;
  }
}

// ---------------------------------------------------------------------------
// Stage 3: deterministically fold the 64 tile-partials per row / column.
// ---------------------------------------------------------------------------
__global__ __launch_bounds__(256) void reduce_parts(
    const float* __restrict__ rowpart, const float* __restrict__ colpart,
    float* __restrict__ rowsum, float* __restrict__ colsum) {
  const int i = blockIdx.x * 256 + threadIdx.x;
  float r = 0.f, c = 0.f;
  #pragma unroll 4
  for (int p = 0; p < NBLK; ++p) {
    r += rowpart[(size_t)p * NROWS + i];
    c += colpart[(size_t)p * NROWS + i];
  }
  rowsum[i] = r;
  colsum[i] = c;
}

// ---------------------------------------------------------------------------
// Stage 4: scalar combine.
// ---------------------------------------------------------------------------
__global__ __launch_bounds__(256) void finalize(
    const float* __restrict__ rowsum, const float* __restrict__ colsum,
    const float* __restrict__ cospair, const float* __restrict__ msq,
    float* __restrict__ out) {
  const int t = threadIdx.x;
  float accC = 0.f, accCos = 0.f, accM = 0.f;
  for (int i = t; i < NROWS; i += 256) {
    const float cp = cospair[i];
    accC  += 0.5f * logf(rowsum[i]) + 0.5f * logf(colsum[i]) - INV_T * cp;
    accCos += 1.0f - cp;
    accM  += msq[i];
  }
  #pragma unroll
  for (int m = 16; m >= 1; m >>= 1) {
    accC  += __shfl_xor(accC, m, 32);
    accCos += __shfl_xor(accCos, m, 32);
    accM  += __shfl_xor(accM, m, 32);
  }
  __shared__ float red[8][3];
  const int lane = t & 31, wid = t >> 5;
  if (lane == 0) { red[wid][0] = accC; red[wid][1] = accCos; red[wid][2] = accM; }
  __syncthreads();
  if (t == 0) {
    float C = 0.f, CO = 0.f, M = 0.f;
    #pragma unroll
    for (int w = 0; w < 8; ++w) { C += red[w][0]; CO += red[w][1]; M += red[w][2]; }
    const float invN = 1.0f / (float)NROWS;
    const float contra  = C * invN;            // mean(0.5*lseR + 0.5*lseC - sim_ii)
    const float cos_emb = CO * invN;
    const float mse     = M / ((float)NROWS * (float)DDIM);
    const float mse_cos = 0.5f * mse + 0.5f * cos_emb;
    out[0] = 0.5f * mse_cos + 0.5f * contra;
  }
}

// ---------------------------------------------------------------------------
extern "C" void kernel_launch(void* const* d_in, const int* in_sizes, int n_in,
                              void* d_out, int out_size, void* d_ws, size_t ws_size,
                              hipStream_t stream) {
  const float* yp = (const float*)d_in[0];
  const float* yy = (const float*)d_in[1];

  // Workspace layout (~36.2 MB total):
  char* ws = (char*)d_ws;
  __bf16* abf = (__bf16*)ws;                                       // 16 MB
  __bf16* bbf = (__bf16*)(ws + (size_t)NROWS * DDIM * 2);          // 16 MB
  float* cospair = (float*)(ws + 2 * (size_t)NROWS * DDIM * 2);    // 32 KB
  float* msq     = cospair + NROWS;                                // 32 KB
  float* rowsum  = msq + NROWS;                                    // 32 KB
  float* colsum  = rowsum + NROWS;                                 // 32 KB
  float* rowpart = colsum + NROWS;                                 // 2 MB
  float* colpart = rowpart + (size_t)NBLK * NROWS;                 // 2 MB

  stage1_rows<<<NROWS, 256, 0, stream>>>(yp, yy, abf, bbf, cospair, msq);
  dim3 grid(NBLK, NBLK);
  gemm_lse<<<grid, 256, 0, stream>>>(abf, bbf, rowpart, colpart);
  reduce_parts<<<NROWS / 256, 256, 0, stream>>>(rowpart, colpart, rowsum, colsum);
  finalize<<<1, 256, 0, stream>>>(rowsum, colsum, cospair, msq, (float*)d_out);
}
